// TransformerBlock_19902878449758
// MI455X (gfx1250) — matrix operand order repaired and checked by reading, not compiled, not run on hardware
//
#include <hip/hip_runtime.h>
#include <hip/hip_bf16.h>
#include <math.h>

// ---------------------------------------------------------------------------
// Transformer block for MI455X (gfx1250, wave32, WMMA 16x16x32 bf16)
// B=2, T=2048, D=1024, H=16, HS=64  ->  M = B*T = 4096, Dff = 4096
//
// Design notes:
//  * All GEMMs run on v_wmma_f32_16x16x32_bf16 (f32 accumulate).
//  * Weights repacked once per launch to fragment-native [(K/16), N, 16]:
//    each WMMA B fragment = one contiguous lane-coalesced 32B global load;
//    GEMM inner loop has no LDS and no barriers (weights live in 192MB L2).
//  * V projection output written transposed [B,H,HS,T] so attention's P@V
//    B-fragments are contiguous global loads.
//  * Attention K tiles staged global->LDS by the Tensor Data Mover
//    (tensor_load_to_lds, TENSORcnt), one DMA per 32-key step.
//  * Attention output tile staged through LDS and written with
//    global_store_async_from_lds_b128 (ASYNCcnt), fully coalesced.
//  * Softmax row-max: DPP ROW_XMASK butterflies (VALU, no LDS);
//    softmax row-sum: on the matrix pipe as P @ ones.
// ---------------------------------------------------------------------------

#define BB  2
#define TT  2048
#define DD  1024
#define HH  16
#define HS  64
#define MM  (BB * TT)     // 4096
#define DFF (4 * DD)      // 4096
#define EPSF 1e-5f

typedef __bf16 bf16;
typedef __bf16 bf16x16 __attribute__((ext_vector_type(16)));
typedef __bf16 bf16x8  __attribute__((ext_vector_type(8)));
typedef __bf16 bf16x4  __attribute__((ext_vector_type(4)));
typedef float  f32x8   __attribute__((ext_vector_type(8)));
typedef unsigned int u32x4 __attribute__((ext_vector_type(4)));
typedef int          i32x8 __attribute__((ext_vector_type(8)));
typedef int          i32x4 __attribute__((ext_vector_type(4)));

// DPP ROW_XMASK butterfly step within a 16-lane row (ctrl = 0x160 | xor_mask)
#define DPP_XOR_F(x, m)                                                        \
    __int_as_float(__builtin_amdgcn_update_dpp(__float_as_int(x),              \
                                               __float_as_int(x),              \
                                               0x160 | (m), 0xF, 0xF, false))

static __device__ __forceinline__ float rowmax16(float x) {
    x = fmaxf(x, DPP_XOR_F(x, 1));
    x = fmaxf(x, DPP_XOR_F(x, 2));
    x = fmaxf(x, DPP_XOR_F(x, 4));
    x = fmaxf(x, DPP_XOR_F(x, 8));
    return x;
}
static __device__ __forceinline__ float rowsum16(float x) {
    x += DPP_XOR_F(x, 1);
    x += DPP_XOR_F(x, 2);
    x += DPP_XOR_F(x, 4);
    x += DPP_XOR_F(x, 8);
    return x;
}

static __device__ __forceinline__ bf16x16 pack16(bf16x8 lo, bf16x8 hi) {
    bf16x16 r;
#pragma unroll
    for (int i = 0; i < 8; ++i) { r[i] = lo[i]; r[8 + i] = hi[i]; }
    return r;
}

// A-matrix 16x32 bf16 fragment: lane holds row M=lane&15,
// K runs [2*h8, 2*h8+15] (contiguous) where h8 = (lane>=16)*8.
static __device__ __forceinline__ bf16x16 load_afrag(const bf16* rowptr, int h8) {
    bf16x8 r0 = *(const bf16x8*)(rowptr + 2 * h8);
    bf16x8 r1 = *(const bf16x8*)(rowptr + 2 * h8 + 8);
    return pack16(r0, r1);
}

// ---------------------------------------------------------------------------
// Weight repack kernels: f32 [K,N] row-major -> bf16 fragment-native layout
// dst[((k>>4)*N + n)*16 + (k&15)] = src[k*N + n]
// ---------------------------------------------------------------------------
__global__ __launch_bounds__(256) void swz_bf16_kernel(const float* __restrict__ src,
                                                       bf16* __restrict__ dst,
                                                       int nShift) {       // log2(N)
    int idx = blockIdx.x * 256 + threadIdx.x;     // output element index
    int klo  = idx & 15;
    int rem  = idx >> 4;                          // kblk*N + n
    int n    = rem & ((1 << nShift) - 1);
    int kblk = rem >> nShift;
    int k    = kblk * 16 + klo;
    dst[idx] = (bf16)src[((size_t)k << nShift) + n];
}

// Wq/Wk/Wv: [H, D, HS] f32 -> fragment-native bf16 with N = H*HS = D
__global__ __launch_bounds__(256) void swz_hdk_kernel(const float* __restrict__ W,
                                                      bf16* __restrict__ dst) {
    int idx = blockIdx.x * 256 + threadIdx.x;
    int klo  = idx & 15;
    int rem  = idx >> 4;
    int n    = rem & 1023;                        // n = h*64 + s
    int kblk = rem >> 10;
    int d    = kblk * 16 + klo;                   // k dim = D
    int h    = n >> 6;
    int s    = n & 63;
    dst[idx] = (bf16)W[((size_t)h * DD + d) * HS + s];
}

// ---------------------------------------------------------------------------
// LayerNorm: one block (256 threads) per token row of D=1024.
// Row reduction: DPP ROW_XMASK within 16-lane groups, then LDS combine.
// ---------------------------------------------------------------------------
__global__ __launch_bounds__(256) void ln_kernel(const float* __restrict__ x,
                                                 const float* __restrict__ g,
                                                 const float* __restrict__ be,
                                                 float* __restrict__ outF,
                                                 bf16* __restrict__ outB) {
    __shared__ float red[32];
    const int row  = blockIdx.x;
    const int tid  = threadIdx.x;
    const int wave = tid >> 5;
    const int lane = tid & 31;
    const int hx   = lane >> 4;
    const int l15  = lane & 15;
    const float* xr = x + (size_t)row * DD;

    float4 v = *(const float4*)(xr + tid * 4);
    float s  = v.x + v.y + v.z + v.w;
    float ss = v.x * v.x + v.y * v.y + v.z * v.z + v.w * v.w;
    s  = rowsum16(s);
    ss = rowsum16(ss);
    if (l15 == 0) { red[wave * 2 + hx] = s; red[16 + wave * 2 + hx] = ss; }
    __syncthreads();
    if (tid == 0) {
        float a = 0.f, b2 = 0.f;
#pragma unroll
        for (int i = 0; i < 16; ++i) { a += red[i]; b2 += red[16 + i]; }
        red[0] = a; red[16] = b2;
    }
    __syncthreads();
    const float mu  = red[0] * (1.0f / DD);
    const float var = red[16] * (1.0f / DD) - mu * mu;
    const float rs  = rsqrtf(var + EPSF);

    float4 gv = *(const float4*)(g + tid * 4);
    float4 bv = *(const float4*)(be + tid * 4);
    float4 o;
    o.x = (v.x - mu) * rs * gv.x + bv.x;
    o.y = (v.y - mu) * rs * gv.y + bv.y;
    o.z = (v.z - mu) * rs * gv.z + bv.z;
    o.w = (v.w - mu) * rs * gv.w + bv.w;
    *(float4*)(outF + (size_t)row * DD + tid * 4) = o;
    bf16x4 ob; ob[0] = (bf16)o.x; ob[1] = (bf16)o.y; ob[2] = (bf16)o.z; ob[3] = (bf16)o.w;
    *(bf16x4*)(outB + (size_t)row * DD + tid * 4) = ob;
}

// ---------------------------------------------------------------------------
// Tiled WMMA GEMM, no LDS: C[M,N] = A_bf16[M,K] @ Bswz[K,N] + bias (+epilogue)
// Block 256 threads = 8 waves (2 in M x 4 in N); wave tile 32x32 (2x2 WMMA);
// block tile 64x128. Bswz is fragment-native (see swz kernels).
// EPI: 0 = bf16 row-major, 1 = exact-GELU bf16, 2 = +res f32,
//      3 = bf16 transposed [B,H,HS,T] (for V)
// ---------------------------------------------------------------------------
template <int EPI>
__global__ __launch_bounds__(256) void gemm_wmma_kernel(
    const bf16* __restrict__ A, const bf16* __restrict__ Bswz,
    const float* __restrict__ bias, const float* __restrict__ res,
    float* __restrict__ outF, bf16* __restrict__ outB,
    int Kdim, int Ndim) {
    const int tid  = threadIdx.x;
    const int lane = tid & 31;
    const int wave = tid >> 5;
    const int hx   = lane >> 4;
    const int l15  = lane & 15;
    const int h8   = hx * 8;

    const int blockN0 = blockIdx.x * 128;
    const int blockM0 = blockIdx.y * 64;
    const int wm = (wave >> 2) * 32;
    const int wn = (wave & 3) * 32;
    const int n0 = blockN0 + wn + l15;            // this lane's B column

    f32x8 acc[2][2] = {};

    for (int k0 = 0; k0 < Kdim; k0 += 32) {
        // A fragments: two contiguous 16B runs per lane, straight from global
        bf16x16 af[2];
#pragma unroll
        for (int mi = 0; mi < 2; ++mi) {
            const int row = blockM0 + wm + mi * 16 + l15;
            const bf16* ap = A + (size_t)row * Kdim + k0;
            if (k0 + 32 < Kdim) __builtin_prefetch(ap + 32, 0, 1);
            af[mi] = load_afrag(ap, h8);
        }
        // B fragments: fragment-native layout -> one 32B load per lane
        const size_t kb = (size_t)((k0 >> 4) + hx) * Ndim;
#pragma unroll
        for (int ni = 0; ni < 2; ++ni) {
            bf16x16 bfrag = *(const bf16x16*)(Bswz + (kb + n0 + ni * 16) * 16);
#pragma unroll
            for (int mi = 0; mi < 2; ++mi) {
                acc[mi][ni] = __builtin_amdgcn_wmma_f32_16x16x32_bf16(
                    false, af[mi], false, bfrag, (short)0, acc[mi][ni], false, false);
            }
        }
    }

    // epilogue
#pragma unroll
    for (int mi = 0; mi < 2; ++mi) {
#pragma unroll
        for (int ni = 0; ni < 2; ++ni) {
            const int ncol = blockN0 + wn + ni * 16 + l15;
            const float bv = bias[ncol];
#pragma unroll
            for (int r = 0; r < 8; ++r) {
                const int row = blockM0 + wm + mi * 16 + r + 8 * hx;
                float v = acc[mi][ni][r] + bv;
                if (EPI == 2) {
                    outF[(size_t)row * Ndim + ncol] = v + res[(size_t)row * Ndim + ncol];
                } else if (EPI == 1) {
                    float gl = 0.5f * v * (1.0f + erff(v * 0.70710678118654752f));
                    outB[(size_t)row * Ndim + ncol] = (bf16)gl;
                } else if (EPI == 3) {
                    // transposed store for V: [B,H,HS,T]
                    const int bb = row >> 11;              // row / TT
                    const int t  = row & (TT - 1);
                    const int hh = ncol >> 6;
                    const int d  = ncol & 63;
                    outB[(((size_t)bb * HH + hh) * HS + d) * TT + t] = (bf16)v;
                } else {
                    outB[(size_t)row * Ndim + ncol] = (bf16)v;
                }
            }
        }
    }
}

// ---------------------------------------------------------------------------
// Flash-attention: block = (b, h, 128 query rows), 8 waves x 16 rows.
// K tiles staged by the TDM (tensor_load_to_lds, one DMA per step);
// V fragments read straight from the transposed V buffer; P transposed
// through wave-private LDS; row-sum via the P @ ones WMMA; output tile
// assembled in LDS and stored with global_store_async_from_lds_b128.
// ---------------------------------------------------------------------------
__global__ __launch_bounds__(256) void attn_kernel(const bf16* __restrict__ qbf,
                                                   const bf16* __restrict__ kbf,
                                                   const bf16* __restrict__ vT,
                                                   bf16* __restrict__ obf) {
    __shared__ __align__(128) bf16 Klds[32][64];      // [s][hs]    4 KB
    __shared__ __align__(128) bf16 Plds[8][16][32];   // per-wave P 8 KB
    __shared__ __align__(128) bf16 Olds[8][16][64];   // per-wave O 16 KB

    const int tid  = threadIdx.x;
    const int lane = tid & 31;
    const int wave = tid >> 5;
    const int hx   = lane >> 4;
    const int l15  = lane & 15;
    const int h8   = hx * 8;

    const int qt = blockIdx.x;     // 128-row q tile
    const int hh = blockIdx.y;     // head
    const int b  = blockIdx.z;     // batch

    // Q fragments (16 rows x 64 hs = two 16x32 A-frags), kept in registers
    const int qrow = qt * 128 + wave * 16 + l15;
    const bf16* qbase = qbf + ((size_t)(b * TT + qrow)) * DD + hh * HS;
    bf16x16 qf[2];
#pragma unroll
    for (int c = 0; c < 2; ++c) qf[c] = load_afrag(qbase + c * 32, h8);

    // ones B-matrix for row-sum-by-WMMA
    bf16x16 ones;
#pragma unroll
    for (int i = 0; i < 16; ++i) ones[i] = (bf16)1.0f;

    f32x8 oacc[4] = {};
    f32x8 lacc = {};
    float mrow[8];
#pragma unroll
    for (int r = 0; r < 8; ++r) mrow[r] = -1e30f;

    const bf16* vTh = vT + ((size_t)(b * HH + hh) * HS) * TT;  // + d*TT + s

    for (int s0 = 0; s0 < TT; s0 += 32) {
        // ---- TDM: stage K tile (32 keys x 64 hs, row stride D) into LDS ----
        if (wave == 0) {
            const unsigned lds_base = (unsigned)(uintptr_t)&Klds[0][0];
            const unsigned long long ga =
                (unsigned long long)(uintptr_t)(kbf + ((size_t)(b * TT + s0)) * DD + hh * HS);
            u32x4 g0;
            g0[0] = 1u;                                        // count=1 (valid D#)
            g0[1] = lds_base;                                  // lds_addr
            g0[2] = (unsigned)(ga & 0xFFFFFFFFu);              // global_addr[31:0]
            g0[3] = ((unsigned)(ga >> 32) & 0x01FFFFFFu)       // global_addr[56:32]
                    | (2u << 30);                              // type=2 ("image")
            i32x8 g1;
            g1[0] = 1 << 16;                                   // data_size=1 -> 2B
            g1[1] = HS << 16;                                  // tensor_dim0[15:0]=64
            g1[2] = 32 << 16;                                  // tensor_dim1[15:0]=32
            g1[3] = HS << 16;                                  // tile_dim0=64
            g1[4] = 32;                                        // tile_dim1=32
            g1[5] = DD;                                        // tensor_dim0_stride=1024
            g1[6] = 0;
            g1[7] = 0;
            i32x4 gz4 = {0, 0, 0, 0};
            i32x8 gz8 = {0, 0, 0, 0, 0, 0, 0, 0};
            __builtin_amdgcn_tensor_load_to_lds(g0, g1, gz4, gz4, gz8, 0);
            __builtin_amdgcn_s_wait_tensorcnt(0);
        }
        __syncthreads();

        // ---- S = Q K^T for 32 keys: two 16x16 C tiles ----
        f32x8 S[2];
#pragma unroll
        for (int st = 0; st < 2; ++st) {
            f32x8 s = {};
#pragma unroll
            for (int c = 0; c < 2; ++c) {
                bf16x16 bfrag = *(const bf16x16*)&Klds[st * 16 + l15][c * 32 + hx * 16];
                s = __builtin_amdgcn_wmma_f32_16x16x32_bf16(
                    false, qf[c], false, bfrag, (short)0, s, false, false);
            }
            S[st] = s;
        }

        // ---- online softmax: DPP row-max, rescale, write P (wave-private) ----
#pragma unroll
        for (int r = 0; r < 8; ++r) {
            float s0v = S[0][r] * 0.125f;     // HS^-0.5
            float s1v = S[1][r] * 0.125f;
            float mx = rowmax16(fmaxf(s0v, s1v));
            float mn = fmaxf(mrow[r], mx);
            float al = __expf(mrow[r] - mn);
            mrow[r] = mn;
            float p0 = __expf(s0v - mn);
            float p1 = __expf(s1v - mn);
            lacc[r] *= al;
#pragma unroll
            for (int dc = 0; dc < 4; ++dc) oacc[dc][r] *= al;
            Plds[wave][r + 8 * hx][l15]      = (bf16)p0;
            Plds[wave][r + 8 * hx][16 + l15] = (bf16)p1;
        }
        // same-wave DS ordering; compiler inserts s_wait_dscnt, no barrier needed
        bf16x16 pf = pack16(*(const bf16x8*)&Plds[wave][l15][2 * h8],
                            *(const bf16x8*)&Plds[wave][l15][2 * h8 + 8]);

        // row-sum on the matrix pipe: l += P @ ones
        lacc = __builtin_amdgcn_wmma_f32_16x16x32_bf16(
            false, pf, false, ones, (short)0, lacc, false, false);

        // ---- O += P @ V : V fragments contiguous from transposed V ----
#pragma unroll
        for (int dc = 0; dc < 4; ++dc) {
            const bf16* vp = vTh + (size_t)(dc * 16 + l15) * TT + s0 + hx * 16;
            if (s0 + 32 < TT) __builtin_prefetch(vp + 32, 0, 1);
            bf16x16 vfrag = *(const bf16x16*)vp;
            oacc[dc] = __builtin_amdgcn_wmma_f32_16x16x32_bf16(
                false, pf, false, vfrag, (short)0, oacc[dc], false, false);
        }
        __syncthreads();   // all waves done with Klds before next TDM issue
    }

    // ---- normalize into wave-private LDS tile, then async store LDS->global ----
#pragma unroll
    for (int r = 0; r < 8; ++r) {
        const float inv = 1.0f / lacc[r];
#pragma unroll
        for (int dc = 0; dc < 4; ++dc) {
            Olds[wave][r + 8 * hx][dc * 16 + l15] = (bf16)(oacc[dc][r] * inv);
        }
    }
    // DScnt and ASYNCcnt are unordered wrt each other: drain LDS writes first
    asm volatile("s_wait_dscnt 0x0" ::: "memory");

    const size_t obase = ((size_t)(b * TT + qt * 128 + wave * 16)) * DD + hh * HS;
#pragma unroll
    for (int i = 0; i < 4; ++i) {
        const int c   = lane + 32 * i;        // 0..127 chunk id (16 rows x 8 chunks)
        const int row = c >> 3;
        const int off = (c & 7) * 8;          // element offset within row
        bf16* gp = obf + obase + (size_t)row * DD + off;
        const unsigned la = (unsigned)(uintptr_t)&Olds[wave][row][off];
        asm volatile("global_store_async_from_lds_b128 %0, %1, off"
                     :: "v"(gp), "v"(la) : "memory");
    }
    asm volatile("s_wait_asynccnt 0x0" ::: "memory");
}

// ---------------------------------------------------------------------------
// Host-side launch
// ---------------------------------------------------------------------------
extern "C" void kernel_launch(void* const* d_in, const int* in_sizes, int n_in,
                              void* d_out, int out_size, void* d_ws, size_t ws_size,
                              hipStream_t stream) {
    (void)in_sizes; (void)n_in; (void)out_size; (void)ws_size;

    const float* x   = (const float*)d_in[0];
    const float* Wq  = (const float*)d_in[1];
    const float* bq  = (const float*)d_in[2];
    const float* Wk  = (const float*)d_in[3];
    const float* bk  = (const float*)d_in[4];
    const float* Wv  = (const float*)d_in[5];
    const float* bv  = (const float*)d_in[6];
    const float* Wp  = (const float*)d_in[7];
    const float* bp  = (const float*)d_in[8];
    const float* W1  = (const float*)d_in[9];
    const float* b1  = (const float*)d_in[10];
    const float* W2  = (const float*)d_in[11];
    const float* b2  = (const float*)d_in[12];
    const float* g1  = (const float*)d_in[13];
    const float* be1 = (const float*)d_in[14];
    const float* g2  = (const float*)d_in[15];
    const float* be2 = (const float*)d_in[16];
    float* out = (float*)d_out;

    // workspace carve-up (256B aligned)
    char* w = (char*)d_ws;
    size_t off = 0;
    auto carve = [&](size_t bytes) {
        char* p = w + off;
        off += (bytes + 255) & ~(size_t)255;
        return p;
    };
    float* h     = (float*)carve((size_t)MM * DD * 4);
    bf16*  hbf   = (bf16*) carve((size_t)MM * DD * 2);
    bf16*  qbf   = (bf16*) carve((size_t)MM * DD * 2);
    bf16*  kbf   = (bf16*) carve((size_t)MM * DD * 2);
    bf16*  vT    = (bf16*) carve((size_t)MM * DD * 2);   // [B,H,HS,T]
    bf16*  obf   = (bf16*) carve((size_t)MM * DD * 2);
    float* r1    = (float*)carve((size_t)MM * DD * 4);
    float* h2n   = (float*)carve((size_t)MM * DD * 4);
    bf16*  h2nbf = (bf16*) carve((size_t)MM * DD * 2);
    bf16*  midbf = (bf16*) carve((size_t)MM * DFF * 2);
    bf16*  Wqt   = (bf16*) carve((size_t)DD * DD * 2);
    bf16*  Wkt   = (bf16*) carve((size_t)DD * DD * 2);
    bf16*  Wvt   = (bf16*) carve((size_t)DD * DD * 2);
    bf16*  Wpt   = (bf16*) carve((size_t)DD * DD * 2);
    bf16*  W1t   = (bf16*) carve((size_t)DD * DFF * 2);
    bf16*  W2t   = (bf16*) carve((size_t)DFF * DD * 2);

    const dim3 blk(256);

    // 1) weight conversion / repack into fragment-native layout
    swz_hdk_kernel<<<(DD * DD) / 256, blk, 0, stream>>>(Wq, Wqt);
    swz_hdk_kernel<<<(DD * DD) / 256, blk, 0, stream>>>(Wk, Wkt);
    swz_hdk_kernel<<<(DD * DD) / 256, blk, 0, stream>>>(Wv, Wvt);
    swz_bf16_kernel<<<(DD * DD) / 256, blk, 0, stream>>>(Wp, Wpt, 10);
    swz_bf16_kernel<<<(DD * DFF) / 256, blk, 0, stream>>>(W1, W1t, 12);
    swz_bf16_kernel<<<(DFF * DD) / 256, blk, 0, stream>>>(W2, W2t, 10);

    // 2) LN1: h = LN(x)*g1+be1 (f32 + bf16)
    ln_kernel<<<MM, blk, 0, stream>>>(x, g1, be1, h, hbf);

    // 3) Q/K/V projections: [4096,1024] @ [1024,1024]
    {
        dim3 grid(DD / 128, MM / 64);
        gemm_wmma_kernel<0><<<grid, blk, 0, stream>>>(hbf, Wqt, bq, nullptr, nullptr, qbf, DD, DD);
        gemm_wmma_kernel<0><<<grid, blk, 0, stream>>>(hbf, Wkt, bk, nullptr, nullptr, kbf, DD, DD);
        gemm_wmma_kernel<3><<<grid, blk, 0, stream>>>(hbf, Wvt, bv, nullptr, nullptr, vT, DD, DD);
    }

    // 4) attention -> obf [B,T,D] bf16
    {
        dim3 grid(TT / 128, HH, BB);
        attn_kernel<<<grid, blk, 0, stream>>>(qbf, kbf, vT, obf);
    }

    // 5) r1 = h + obf @ Wp + bp   (f32)
    {
        dim3 grid(DD / 128, MM / 64);
        gemm_wmma_kernel<2><<<grid, blk, 0, stream>>>(obf, Wpt, bp, h, r1, nullptr, DD, DD);
    }

    // 6) LN2: h2n = LN(r1)*g2+be2 (f32 + bf16)
    ln_kernel<<<MM, blk, 0, stream>>>(r1, g2, be2, h2n, h2nbf);

    // 7) FFN1: mid = gelu(h2n @ W1 + b1)  bf16 [4096,4096]
    {
        dim3 grid(DFF / 128, MM / 64);
        gemm_wmma_kernel<1><<<grid, blk, 0, stream>>>(h2nbf, W1t, b1, nullptr, nullptr, midbf, DD, DFF);
    }

    // 8) out = h2n + mid @ W2 + b2  (f32 -> d_out)
    {
        dim3 grid(DD / 128, MM / 64);
        gemm_wmma_kernel<2><<<grid, blk, 0, stream>>>(midbf, W2t, b2, h2n, out, nullptr, DFF, DD);
    }
}
